// TransformerEncoder_4629974745570
// MI455X (gfx1250) — matrix operand order repaired and checked by reading, not compiled, not run on hardware
//
#include <hip/hip_runtime.h>
#include <cstdint>
#include <cstddef>

// ---------------------------------------------------------------------------
// Problem constants (match reference)
// ---------------------------------------------------------------------------
constexpr int Bn   = 4;          // batch
constexpr int Sn   = 1024;       // sequence
constexpr int Dn   = 1024;       // model dim
constexpr int Hn   = 16;         // heads
constexpr int DHn  = 64;         // head dim
constexpr int Ln   = 4;          // layers
constexpr int MLPn = 4096;       // mlp dim
constexpr int BS   = Bn * Sn;    // 4096 rows

// ---------------------------------------------------------------------------
// CDNA5 WMMA types / helpers (wave32, v_wmma_f32_16x16x32_bf16)
// ---------------------------------------------------------------------------
typedef __bf16  v16bf __attribute__((ext_vector_type(16)));
typedef float   v8f   __attribute__((ext_vector_type(8)));
typedef unsigned int u32x4 __attribute__((ext_vector_type(4)));

union BF16Frag { u32x4 q[2]; v16bf v; __bf16 e[16]; };

__device__ inline v8f wmma_bf16(v16bf a, v16bf b, v8f c) {
  // D = A(16x32) * B(32x16) + C(16x16 f32)
  return __builtin_amdgcn_wmma_f32_16x16x32_bf16(
      /*neg_a=*/false, a, /*neg_b=*/false, b,
      /*c_mod=*/(short)0, c, /*reuse_a=*/false, /*reuse_b=*/false);
}

// A-matrix fragment (16x32 bf16): lane16 = row; per-lane 16 elems in two 16B
// chunks at K offsets {hi*8..} and {16+hi*8..} (ISA 7.12.2).
__device__ inline v16bf ld_a_frag(const __bf16* rowbase, int hi) {
  BF16Frag u;
  u.q[0] = *reinterpret_cast<const u32x4*>(rowbase + hi * 8);
  u.q[1] = *reinterpret_cast<const u32x4*>(rowbase + 16 + hi * 8);
  return u.v;
}

// B-matrix fragment (32x16 bf16): lane16 = column; per-lane 16 K values in two
// 16B chunks at K offsets {hi*8..} and {16+hi*8..} (contraction-contiguous storage).
__device__ inline v16bf ld_b_frag(const __bf16* colbase, int hi) {
  BF16Frag u;
  u.q[0] = *reinterpret_cast<const u32x4*>(colbase + hi * 8);
  u.q[1] = *reinterpret_cast<const u32x4*>(colbase + 16 + hi * 8);
  return u.v;
}

// ---------------------------------------------------------------------------
// CDNA5 async copy global -> LDS (GLOBAL_LOAD_ASYNC_TO_LDS_B128, ASYNCcnt).
// Generic pointer to __shared__ truncates to the LDS byte address (ISA 10.2).
// Async loads complete in order; s_wait_asynccnt N retires all but N newest.
// ---------------------------------------------------------------------------
__device__ inline void async_copy_b128(void* lds_dst, const void* gsrc) {
  unsigned l = (unsigned)(size_t)lds_dst;
  asm volatile("global_load_async_to_lds_b128 %0, %1, off"
               :: "v"(l), "v"(gsrc) : "memory");
}
__device__ inline void wait_async_le4() {
  asm volatile("s_wait_asynccnt 4" ::: "memory");
}
__device__ inline void wait_async_le0() {
  asm volatile("s_wait_asynccnt 0" ::: "memory");
}

// ---------------------------------------------------------------------------
// bf16 tiled GEMM:  C(MxN,f32) = A(MxK,bf16,rowmaj) * Bt(NxK,bf16,rowmaj)
//                               + bias[N] (+ residual)
// Weights are pre-transposed so BOTH tiles are contraction-contiguous and
// stream through double-buffered LDS via async b128 copies (4 per thread per
// tile; pipeline paced with s_wait_asynccnt).
// Block: 256 thr = 8 waves; block tile 128x128; wave tile 32x64 (2x4 frags).
// ---------------------------------------------------------------------------
__global__ __launch_bounds__(256) void gemm_bf16_kernel(
    const __bf16* __restrict__ A, const __bf16* __restrict__ Bt,
    const float* __restrict__ bias, const float* __restrict__ resid,
    float* __restrict__ C, int M, int N, int K) {
  constexpr int LDT = 40;                       // LDS row stride (elems)
  __shared__ __bf16 As[2][128 * LDT];           // [row][k]
  __shared__ __bf16 Bs[2][128 * LDT];           // [n][k]

  const int tid = threadIdx.x;
  const int lane = tid & 31, lane16 = lane & 15, hi = lane >> 4;
  const int wid = tid >> 5, wr = wid >> 1, wc = wid & 1;
  const int m0w = wr * 32, n0w = wc * 64;
  const int bm = blockIdx.y * 128, bn = blockIdx.x * 128;

  v8f acc[2][4];
#pragma unroll
  for (int i = 0; i < 2; ++i)
#pragma unroll
    for (int j = 0; j < 4; ++j)
#pragma unroll
      for (int e = 0; e < 8; ++e) acc[i][j][e] = 0.f;

  const int ldrow = tid >> 1, ldhalf = (tid & 1) * 16;  // tile loader map

  auto issue_tile = [&](int k0, int buf) {
    const __bf16* asrc = A + (size_t)(bm + ldrow) * K + k0 + ldhalf;
    __bf16* adst = &As[buf][ldrow * LDT + ldhalf];
    async_copy_b128(adst, asrc);
    async_copy_b128(adst + 8, asrc + 8);
    const __bf16* bsrc = Bt + (size_t)(bn + ldrow) * K + k0 + ldhalf;
    __bf16* bdst = &Bs[buf][ldrow * LDT + ldhalf];
    async_copy_b128(bdst, bsrc);
    async_copy_b128(bdst + 8, bsrc + 8);
  };

  const int KT = K / 32;
  issue_tile(0, 0);
  for (int kt = 0; kt < KT; ++kt) {
    const int buf = kt & 1;
    if (kt + 1 < KT) {
      issue_tile((kt + 1) * 32, buf ^ 1);  // prefetch next tile
      wait_async_le4();                    // retire current tile's 4 copies
    } else {
      wait_async_le0();
    }
    __syncthreads();

    v16bf a0 = ld_a_frag(&As[buf][(m0w + lane16) * LDT], hi);
    v16bf a1 = ld_a_frag(&As[buf][(m0w + 16 + lane16) * LDT], hi);
#pragma unroll
    for (int cf = 0; cf < 4; ++cf) {
      v16bf bf = ld_b_frag(&Bs[buf][(n0w + cf * 16 + lane16) * LDT], hi);
      acc[0][cf] = wmma_bf16(a0, bf, acc[0][cf]);
      acc[1][cf] = wmma_bf16(a1, bf, acc[1][cf]);
    }
    __syncthreads();  // all reads done before buf is overwritten again
  }

  // Epilogue: C/D layout — lane16 = col, VGPR r -> row hi*8+r (ISA 7.12.2)
#pragma unroll
  for (int rf = 0; rf < 2; ++rf)
#pragma unroll
    for (int cf = 0; cf < 4; ++cf)
#pragma unroll
      for (int r = 0; r < 8; ++r) {
        int row = bm + m0w + rf * 16 + hi * 8 + r;
        int col = bn + n0w + cf * 16 + lane16;
        float v = acc[rf][cf][r] + bias[col];
        if (resid) v += resid[(size_t)row * N + col];
        C[(size_t)row * N + col] = v;
      }
}

// ---------------------------------------------------------------------------
// Flash-style attention. Q/K laid out [B][H][S][DH] bf16, V transposed
// [B][H][DH][S] bf16 (scales folded in at RoPE). Double-buffered async K/V
// tiles; block = 256 thr handles one (b,h,128-query slab); wave owns 16 rows.
// ---------------------------------------------------------------------------
__global__ __launch_bounds__(256) void attn_kernel(
    const __bf16* __restrict__ Q, const __bf16* __restrict__ Kh,
    const __bf16* __restrict__ Vt, __bf16* __restrict__ Y) {
  constexpr int KST = 72, PST = 40;
  __shared__ __bf16 Ks[2][64 * KST];     // [key][dh]
  __shared__ __bf16 Vs[2][64 * KST];     // [dh][key]
  __shared__ __bf16 Ps[8][16 * PST];     // per-wave P tile (16x32)

  const int tid = threadIdx.x, lane = tid & 31, lane16 = lane & 15,
            hi = tid >> 4 & 1, w = tid >> 5;
  const int bh = blockIdx.x >> 3, qblk = blockIdx.x & 7;
  const int q0 = qblk * 128 + w * 16;
  const size_t basebh = (size_t)bh * Sn * DHn;

  v16bf aq[2];
#pragma unroll
  for (int dhc = 0; dhc < 2; ++dhc)
    aq[dhc] = ld_a_frag(Q + basebh + (size_t)(q0 + lane16) * DHn + dhc * 32, hi);

  v8f accv[4];
#pragma unroll
  for (int f = 0; f < 4; ++f)
#pragma unroll
    for (int e = 0; e < 8; ++e) accv[f][e] = 0.f;
  float mrow[8], lrow[8];
#pragma unroll
  for (int r = 0; r < 8; ++r) { mrow[r] = -1e30f; lrow[r] = 0.f; }

  const int ld_row = tid >> 2, ld_c = (tid & 3) * 16;

  auto issue_kv = [&](int kt, int buf) {
    const __bf16* ksrc = Kh + basebh + (size_t)(kt + ld_row) * DHn + ld_c;
    __bf16* kdst = &Ks[buf][ld_row * KST + ld_c];
    async_copy_b128(kdst, ksrc);
    async_copy_b128(kdst + 8, ksrc + 8);
    const __bf16* vsrc = Vt + basebh + (size_t)ld_row * Sn + kt + ld_c;
    __bf16* vdst = &Vs[buf][ld_row * KST + ld_c];
    async_copy_b128(vdst, vsrc);
    async_copy_b128(vdst + 8, vsrc + 8);
  };

  constexpr int NKT = Sn / 64;
  issue_kv(0, 0);
  for (int ktile = 0; ktile < NKT; ++ktile) {
    const int buf = ktile & 1;
    if (ktile + 1 < NKT) {
      issue_kv((ktile + 1) * 64, buf ^ 1);
      wait_async_le4();
    } else {
      wait_async_le0();
    }
    __syncthreads();

#pragma unroll
    for (int kb = 0; kb < 2; ++kb) {   // two 32-key sub-blocks
      v8f s0, s1;
#pragma unroll
      for (int e = 0; e < 8; ++e) { s0[e] = 0.f; s1[e] = 0.f; }
#pragma unroll
      for (int dhc = 0; dhc < 2; ++dhc) {
        v16bf b0 = ld_b_frag(&Ks[buf][(kb * 32 + lane16) * KST + dhc * 32], hi);
        v16bf b1 =
            ld_b_frag(&Ks[buf][(kb * 32 + 16 + lane16) * KST + dhc * 32], hi);
        s0 = wmma_bf16(aq[dhc], b0, s0);
        s1 = wmma_bf16(aq[dhc], b1, s1);
      }
      // online softmax (row r lives in lanes hi*16..hi*16+15; xor<16 stays
      // within the half on wave32)
#pragma unroll
      for (int r = 0; r < 8; ++r) {
        float bmx = fmaxf(s0[r], s1[r]);
        bmx = fmaxf(bmx, __shfl_xor(bmx, 1));
        bmx = fmaxf(bmx, __shfl_xor(bmx, 2));
        bmx = fmaxf(bmx, __shfl_xor(bmx, 4));
        bmx = fmaxf(bmx, __shfl_xor(bmx, 8));
        float mnew = fmaxf(mrow[r], bmx);
        float sc = __expf(mrow[r] - mnew);
        float p0 = __expf(s0[r] - mnew);
        float p1 = __expf(s1[r] - mnew);
        float rs = p0 + p1;
        rs += __shfl_xor(rs, 1);
        rs += __shfl_xor(rs, 2);
        rs += __shfl_xor(rs, 4);
        rs += __shfl_xor(rs, 8);
        lrow[r] = lrow[r] * sc + rs;
        mrow[r] = mnew;
#pragma unroll
        for (int f = 0; f < 4; ++f) accv[f][r] *= sc;
        Ps[w][(hi * 8 + r) * PST + lane16] = (__bf16)p0;
        Ps[w][(hi * 8 + r) * PST + 16 + lane16] = (__bf16)p1;
      }
      __syncthreads();
      v16bf ap = ld_a_frag(&Ps[w][lane16 * PST], hi);
#pragma unroll
      for (int f = 0; f < 4; ++f) {
        v16bf bv = ld_b_frag(&Vs[buf][(f * 16 + lane16) * KST + kb * 32], hi);
        accv[f] = wmma_bf16(ap, bv, accv[f]);
      }
      __syncthreads();
    }
  }

  // Epilogue: write Y as [b][s][h*DH+dh] bf16 for the out-proj GEMM.
  const int b = bh >> 4, h = bh & 15;
#pragma unroll
  for (int r = 0; r < 8; ++r) {
    float inv = 1.f / lrow[r];
    int srow = q0 + hi * 8 + r;
    size_t base = ((size_t)(b * Sn + srow)) * Dn + h * DHn;
#pragma unroll
    for (int f = 0; f < 4; ++f)
      Y[base + f * 16 + lane16] = (__bf16)(accv[f][r] * inv);
  }
}

// ---------------------------------------------------------------------------
// LayerNorm (row = 1024). Writes bf16 and/or f32 outputs.
// ---------------------------------------------------------------------------
__global__ __launch_bounds__(256) void ln_kernel(
    const float* __restrict__ x, const float* __restrict__ w,
    const float* __restrict__ b, __bf16* __restrict__ ob,
    float* __restrict__ of) {
  __shared__ float r1[256], r2[256];
  const int row = blockIdx.x, t = threadIdx.x;
  const float* xr = x + (size_t)row * Dn;
  float v0[4], s = 0.f, sq = 0.f;
#pragma unroll
  for (int j = 0; j < 4; ++j) {
    v0[j] = xr[t + j * 256];
    s += v0[j];
    sq += v0[j] * v0[j];
  }
  r1[t] = s; r2[t] = sq;
  __syncthreads();
  for (int o = 128; o > 0; o >>= 1) {
    if (t < o) { r1[t] += r1[t + o]; r2[t] += r2[t + o]; }
    __syncthreads();
  }
  float mean = r1[0] * (1.f / Dn);
  float var = r2[0] * (1.f / Dn) - mean * mean;
  float rstd = rsqrtf(var + 1e-5f);
#pragma unroll
  for (int j = 0; j < 4; ++j) {
    int i = t + j * 256;
    float val = (v0[j] - mean) * rstd * w[i] + b[i];
    if (ob) ob[(size_t)row * Dn + i] = (__bf16)val;
    if (of) of[(size_t)row * Dn + i] = val;
  }
}

// ---------------------------------------------------------------------------
// RoPE(XPos): qkv f32 [BS][3*D] -> q/k bf16 [B][H][S][DH], v bf16 [B][H][DH][S]
// Folds attn scale DH^-0.5 + xpos scale into q, 1/scale into k.
// ---------------------------------------------------------------------------
__global__ __launch_bounds__(256) void rope_kernel(
    const float* __restrict__ qkv, __bf16* __restrict__ qo,
    __bf16* __restrict__ ko, __bf16* __restrict__ vt) {
  int p = blockIdx.x * blockDim.x + threadIdx.x;   // 2^21 threads
  int d2 = p & 31, h = (p >> 5) & 15, s = (p >> 9) & 1023, bb = p >> 19;
  size_t ib = (size_t)(bb * Sn + s) * (3 * Dn) + h * DHn + 2 * d2;
  float q0 = qkv[ib], q1 = qkv[ib + 1];
  float k0 = qkv[ib + Dn], k1 = qkv[ib + Dn + 1];
  float vv0 = qkv[ib + 2 * Dn], vv1 = qkv[ib + 2 * Dn + 1];
  float inv = __powf(10000.f, -(float)(2 * d2) * (1.f / DHn));
  float f = (float)s * inv;
  float c = __cosf(f), sn = __sinf(f);
  float basev = ((float)(2 * d2) + 0.4f * DHn) / (1.4f * DHn);
  float power = ((float)s - (float)(Sn / 2)) * (1.f / 512.f);
  float scale = __powf(basev, power);
  float qs = scale * 0.125f;        // * DH^-0.5
  float is = 1.f / scale;
  size_t ob = (size_t)((bb * Hn + h) * Sn + s) * DHn + 2 * d2;
  qo[ob]     = (__bf16)((q0 * c - q1 * sn) * qs);
  qo[ob + 1] = (__bf16)((q1 * c + q0 * sn) * qs);
  ko[ob]     = (__bf16)((k0 * c - k1 * sn) * is);
  ko[ob + 1] = (__bf16)((k1 * c + k0 * sn) * is);
  size_t ovt = ((size_t)(bb * Hn + h) * DHn + 2 * d2) * Sn + s;  // transposed
  vt[ovt]      = (__bf16)vv0;
  vt[ovt + Sn] = (__bf16)vv1;
}

// ---------------------------------------------------------------------------
// Elementwise kernels
// ---------------------------------------------------------------------------
__global__ void silumul_kernel(const float* __restrict__ t1,
                               const float* __restrict__ t2,
                               __bf16* __restrict__ o, size_t n) {
  size_t i = (size_t)blockIdx.x * blockDim.x + threadIdx.x;
  size_t stride = (size_t)gridDim.x * blockDim.x;
  for (; i < n; i += stride) {
    float a = t1[i];
    float silu = a / (1.f + __expf(-a));
    o[i] = (__bf16)(silu * t2[i]);
  }
}

// fp32 [L][K][N] -> bf16 [L][N][K] (weights pre-transposed so GEMM B tiles
// are contraction-contiguous for async-to-LDS streaming)
__global__ void cvt_bf16_t_kernel(const float* __restrict__ in,
                                  __bf16* __restrict__ out, int K, int N,
                                  size_t total) {
  size_t i = (size_t)blockIdx.x * blockDim.x + threadIdx.x;
  size_t stride = (size_t)gridDim.x * blockDim.x;
  size_t plane = (size_t)K * N;
  for (; i < total; i += stride) {
    size_t l = i / plane;
    size_t rem = i - l * plane;
    int k = (int)(rem / N), n = (int)(rem % N);
    out[l * plane + (size_t)n * K + k] = (__bf16)in[i];
  }
}

__global__ void copy_f32_kernel(const float* __restrict__ in,
                                float* __restrict__ out, size_t n) {
  size_t i = (size_t)blockIdx.x * blockDim.x + threadIdx.x;
  size_t stride = (size_t)gridDim.x * blockDim.x;
  for (; i < n; i += stride) out[i] = in[i];
}

// ---------------------------------------------------------------------------
// Host orchestration
// ---------------------------------------------------------------------------
extern "C" void kernel_launch(void* const* d_in, const int* in_sizes, int n_in,
                              void* d_out, int out_size, void* d_ws,
                              size_t ws_size, hipStream_t stream) {
  (void)in_sizes; (void)n_in; (void)out_size; (void)ws_size;
  const float* x_in  = (const float*)d_in[0];
  const float* ln1_w = (const float*)d_in[1];
  const float* ln1_b = (const float*)d_in[2];
  const float* wqkv  = (const float*)d_in[3];
  const float* bqkv  = (const float*)d_in[4];
  const float* wo    = (const float*)d_in[5];
  const float* bo    = (const float*)d_in[6];
  const float* ln2_w = (const float*)d_in[7];
  const float* ln2_b = (const float*)d_in[8];
  const float* w1    = (const float*)d_in[9];
  const float* b1    = (const float*)d_in[10];
  const float* w2    = (const float*)d_in[11];
  const float* b2    = (const float*)d_in[12];
  const float* wout  = (const float*)d_in[13];
  const float* bout  = (const float*)d_in[14];
  const float* fln_w = (const float*)d_in[15];
  const float* fln_b = (const float*)d_in[16];
  float* out = (float*)d_out;

  // Bump allocator over workspace (all sizes multiple of 256 bytes).
  char* ws = (char*)d_ws;
  size_t off = 0;
  auto alloc = [&](size_t bytes) {
    char* p = ws + off;
    off += (bytes + 255) & ~(size_t)255;
    return p;
  };
  __bf16* wqkv_b = (__bf16*)alloc((size_t)Ln * Dn * 3 * Dn * 2);  // [l][3D][D]
  __bf16* wo_b   = (__bf16*)alloc((size_t)Ln * Dn * Dn * 2);      // [l][D][D]
  __bf16* w1_b   = (__bf16*)alloc((size_t)Ln * Dn * MLPn * 2);    // [l][MLP][D]
  __bf16* w2_b   = (__bf16*)alloc((size_t)Ln * Dn * MLPn * 2);    // [l][MLP][D]
  __bf16* wout_b = (__bf16*)alloc((size_t)Ln * MLPn * Dn * 2);    // [l][D][MLP]
  float*  xbuf   = (float*)alloc((size_t)BS * Dn * 4);
  __bf16* nrm    = (__bf16*)alloc((size_t)BS * Dn * 2);
  float*  qkvf   = (float*)alloc((size_t)BS * 3 * Dn * 4);
  __bf16* qh     = (__bf16*)alloc((size_t)Bn * Hn * Sn * DHn * 2);
  __bf16* kh     = (__bf16*)alloc((size_t)Bn * Hn * Sn * DHn * 2);
  __bf16* vt     = (__bf16*)alloc((size_t)Bn * Hn * Sn * DHn * 2);
  __bf16* ybuf   = (__bf16*)alloc((size_t)BS * Dn * 2);
  float*  t2f    = (float*)alloc((size_t)BS * MLPn * 4);
  __bf16* mmb    = (__bf16*)alloc((size_t)BS * MLPn * 2);
  // t1 (64MB) aliases the dead qkv/q/k region (exactly 64MB) during MLP phase
  float*  t1f    = (float*)qkvf;

  const dim3 blk(256);
  // 0) copy x into workspace; convert + transpose weights to bf16 [N][K]
  copy_f32_kernel<<<4096, blk, 0, stream>>>(x_in, xbuf, (size_t)BS * Dn);
  cvt_bf16_t_kernel<<<4096, blk, 0, stream>>>(wqkv, wqkv_b, Dn, 3 * Dn,
                                              (size_t)Ln * Dn * 3 * Dn);
  cvt_bf16_t_kernel<<<4096, blk, 0, stream>>>(wo, wo_b, Dn, Dn,
                                              (size_t)Ln * Dn * Dn);
  cvt_bf16_t_kernel<<<4096, blk, 0, stream>>>(w1, w1_b, Dn, MLPn,
                                              (size_t)Ln * Dn * MLPn);
  cvt_bf16_t_kernel<<<4096, blk, 0, stream>>>(w2, w2_b, Dn, MLPn,
                                              (size_t)Ln * Dn * MLPn);
  cvt_bf16_t_kernel<<<4096, blk, 0, stream>>>(wout, wout_b, MLPn, Dn,
                                              (size_t)Ln * MLPn * Dn);

  for (int l = 0; l < Ln; ++l) {
    // LN1 -> bf16
    ln_kernel<<<BS, blk, 0, stream>>>(xbuf, ln1_w + l * Dn, ln1_b + l * Dn,
                                      nrm, nullptr);
    // QKV GEMM: [4096x1024] x [1024x3072]
    gemm_bf16_kernel<<<dim3(3 * Dn / 128, BS / 128), blk, 0, stream>>>(
        nrm, wqkv_b + (size_t)l * Dn * 3 * Dn, bqkv + (size_t)l * 3 * Dn,
        nullptr, qkvf, BS, 3 * Dn, Dn);
    // RoPE + head split + scale folding (V emitted transposed)
    rope_kernel<<<(Bn * Sn * Hn * (DHn / 2)) / 256, blk, 0, stream>>>(
        qkvf, qh, kh, vt);
    // Attention (flash-style, WMMA both matmuls, async K/V tiles)
    attn_kernel<<<Bn * Hn * (Sn / 128), blk, 0, stream>>>(qh, kh, vt, ybuf);
    // Out-proj + residual: x = x + y @ wo + bo
    gemm_bf16_kernel<<<dim3(Dn / 128, BS / 128), blk, 0, stream>>>(
        ybuf, wo_b + (size_t)l * Dn * Dn, bo + (size_t)l * Dn, xbuf, xbuf,
        BS, Dn, Dn);
    // LN2 -> bf16
    ln_kernel<<<BS, blk, 0, stream>>>(xbuf, ln2_w + l * Dn, ln2_b + l * Dn,
                                      nrm, nullptr);
    // MLP: t1 = h@w1+b1 ; t2 = h@w2+b2 ; m = silu(t1)*t2 ; x += m@wout+bout
    gemm_bf16_kernel<<<dim3(MLPn / 128, BS / 128), blk, 0, stream>>>(
        nrm, w1_b + (size_t)l * Dn * MLPn, b1 + (size_t)l * MLPn, nullptr,
        t1f, BS, MLPn, Dn);
    gemm_bf16_kernel<<<dim3(MLPn / 128, BS / 128), blk, 0, stream>>>(
        nrm, w2_b + (size_t)l * Dn * MLPn, b2 + (size_t)l * MLPn, nullptr,
        t2f, BS, MLPn, Dn);
    silumul_kernel<<<8192, blk, 0, stream>>>(t1f, t2f, mmb,
                                             (size_t)BS * MLPn);
    gemm_bf16_kernel<<<dim3(Dn / 128, BS / 128), blk, 0, stream>>>(
        mmb, wout_b + (size_t)l * MLPn * Dn, bout + (size_t)l * Dn, xbuf,
        xbuf, BS, Dn, MLPn);
  }
  // Final LN -> fp32 output
  ln_kernel<<<BS, blk, 0, stream>>>(xbuf, fln_w, fln_b, nullptr, out);
}
